// RelationalReasoningModule_14645838479606
// MI455X (gfx1250) — matrix operand order repaired and checked by reading, not compiled, not run on hardware
//
#include <hip/hip_runtime.h>
#include <hip/hip_bf16.h>

typedef __attribute__((ext_vector_type(16))) _Float16 v16h;
typedef __attribute__((ext_vector_type(8)))  _Float16 v8h;
typedef __attribute__((ext_vector_type(8)))  float    v8f;

#define BATCH 64
#define NOBJ  64
#define DIM   256
#define DIM2  512
#define OUTD  256
#define NPAIR 2016
#define PTILE 16
#define NBLK  (NPAIR / PTILE)   // 126 pair-tiles per batch
#define LNEPS 1e-5f
#define SBN   (DIM * 32)        // halves per weight K-slice buffer (16 KB)

// ---------------- CDNA5 async copy helpers ----------------
// GLOBAL_LOAD_ASYNC_TO_LDS_B128: per-lane 16B global->LDS, tracked by ASYNCcnt.
__device__ __forceinline__ void async_b128(unsigned lds_off, const void* g) {
    asm volatile("global_load_async_to_lds_b128 %0, %1, off"
                 :: "v"(lds_off), "v"((unsigned long long)(uintptr_t)g)
                 : "memory");
}
__device__ __forceinline__ void wait_async_le4() {
    asm volatile("s_wait_asynccnt 0x4" ::: "memory");
}
__device__ __forceinline__ void wait_async_0() {
    asm volatile("s_wait_asynccnt 0x0" ::: "memory");
}

// Stage one 32-wide K-slice of a transposed weight matrix [256][rowHalves]
// into LDS as [256 rows][32 halves]; thread t copies row t (64 B = 4 async b128).
__device__ __forceinline__ void stage_wslice(const _Float16* __restrict__ Wt, int rowHalves,
                                             int kk, _Float16* sBuf, int t) {
    const char* src = (const char*)(Wt + (size_t)t * rowHalves + kk * 32);
    unsigned dst = (unsigned)(uintptr_t)(sBuf + t * 32);
    #pragma unroll
    for (int q = 0; q < 4; ++q)
        async_b128(dst + 16u * q, src + 16 * q);
}

// ---------------- prep kernels ----------------
__global__ void k_f2h(const float* __restrict__ src, _Float16* __restrict__ dst, int n) {
    int i = blockIdx.x * blockDim.x + threadIdx.x;
    if (i < n) dst[i] = (_Float16)src[i];
}

// W is [K][Ncol] row-major; Wt is [Ncol][K] (f16) so B-fragments read K-contiguous.
__global__ void k_transpose_h(const float* __restrict__ W, _Float16* __restrict__ Wt,
                              int K, int Ncol) {
    int i = blockIdx.x * blockDim.x + threadIdx.x;
    if (i < K * Ncol) {
        int k = i / Ncol, n = i % Ncol;
        Wt[(size_t)n * K + k] = (_Float16)W[i];
    }
}

__global__ void k_zero(float* __restrict__ p, int n) {
    int i = blockIdx.x * blockDim.x + threadIdx.x;
    if (i < n) p[i] = 0.f;
}

// ---------------- fused pair-relational kernel ----------------
// One block per (batch, 16-pair tile). 256 threads = 8 waves.
// Wave w owns output columns [32w, 32w+32) as two 16x16 WMMA N-tiles.
// Weights stream through a double-buffered LDS K-slice via async-to-LDS.
__launch_bounds__(256, 2)
__global__ void k_pairs(const _Float16* __restrict__ xh,
                        const _Float16* __restrict__ W1t, const float* __restrict__ b1,
                        const float* __restrict__ g1,    const float* __restrict__ be1,
                        const _Float16* __restrict__ W2t, const float* __restrict__ b2,
                        const float* __restrict__ g2,    const float* __restrict__ be2,
                        float* __restrict__ pooled) {
    __shared__ _Float16 sPair[PTILE * DIM2];     // 16 KB: 16 pair rows x 512 f16
    __shared__ _Float16 sB[2 * SBN];             // 32 KB: double-buffered weight K-slices
    __shared__ float    sZ[PTILE * DIM];         // 16 KB: pre-LN activations (f32)
    __shared__ _Float16 sH[PTILE * DIM];         //  8 KB: post-LN/ReLU activations (f16)
    __shared__ float    sPart[PTILE][16][2];     // LN partial sums
    __shared__ float    sStats[PTILE][2];        // per-row mean, rsqrt(var+eps)

    const int t    = threadIdx.x;
    const int lane = t & 31;
    const int wave = t >> 5;
    const int b    = blockIdx.x / NBLK;
    const int pb   = blockIdx.x % NBLK;
    const int pbase = pb * PTILE;

    // ---- async-gather 16 pair rows: concat(x[b,i,:], x[b,j,:]) into LDS ----
    {
        int m = t >> 4;           // pair row within tile
        int chunk = t & 15;       // 32-half (64 B) chunk of the 512-wide row
        int p = pbase + m;
        int i = 0, off = 0;
        while (off + (NOBJ - 1 - i) <= p) { off += (NOBJ - 1 - i); ++i; }
        int j = p - off + i + 1;
        int srow = (chunk < 8) ? i : j;
        int coff = (chunk & 7) * 32;
        const char* src = (const char*)(xh + ((size_t)b * NOBJ + srow) * DIM + coff);
        unsigned dst = (unsigned)(uintptr_t)(&sPair[m * DIM2 + chunk * 32]);
        #pragma unroll
        for (int q = 0; q < 4; ++q)
            async_b128(dst + 16u * q, src + 16 * q);
    }
    // prefetch W1 K-slice 0 behind the gather
    stage_wslice(W1t, DIM2, 0, sB, t);

    const int mrow = lane & 15;   // A row / B column within 16-wide tile
    const int hi   = lane >> 4;   // K half-select (contiguous k = 16*hi + s, same for A and B)
    const int nb0  = wave * 32;
    const int nb1  = wave * 32 + 16;
    const int n0   = nb0 + mrow;
    const int n1   = nb1 + mrow;

    // ================= GEMM1: [16,512] x W1t -> z1 [16,256] =================
    v8f c0 = {0.f,0.f,0.f,0.f,0.f,0.f,0.f,0.f};
    v8f c1 = {0.f,0.f,0.f,0.f,0.f,0.f,0.f,0.f};
    #pragma unroll 2
    for (int kk = 0; kk < DIM2 / 32; ++kk) {
        if (kk + 1 < DIM2 / 32) {                 // stream next weight slice
            stage_wslice(W1t, DIM2, kk + 1, sB + ((kk + 1) & 1) * SBN, t);
            wait_async_le4();                     // slice kk (and gather) landed
        } else {
            wait_async_0();
        }
        __syncthreads();
        const _Float16* sBk = sB + (kk & 1) * SBN;
        const _Float16* ap = &sPair[mrow * DIM2 + kk * 32 + hi * 16];
        v8h alo = *(const v8h*)ap;          // K = kb..kb+7
        v8h ahi = *(const v8h*)(ap + 8);    // K = kb+8..kb+15
        v16h a = __builtin_shufflevector(alo, ahi, 0,1,2,3,4,5,6,7,8,9,10,11,12,13,14,15);
        v16h bf0 = *(const v16h*)(sBk + n0 * 32 + hi * 16);
        v16h bf1 = *(const v16h*)(sBk + n1 * 32 + hi * 16);
        c0 = __builtin_amdgcn_wmma_f32_16x16x32_f16(false, a, false, bf0, (short)0, c0, false, false);
        c1 = __builtin_amdgcn_wmma_f32_16x16x32_f16(false, a, false, bf1, (short)0, c1, false, false);
        __syncthreads();                          // protect buffer being refilled next iter
    }
    // prefetch W2 K-slice 0; it overlaps the LayerNorm phase below
    stage_wslice(W2t, DIM, 0, sB, t);
    {   // + bias, spill to LDS for LayerNorm
        float bb0 = b1[n0], bb1 = b1[n1];
        #pragma unroll
        for (int r = 0; r < 8; ++r) {
            int row = r + 8 * hi;
            sZ[row * DIM + n0] = c0[r] + bb0;
            sZ[row * DIM + n1] = c1[r] + bb1;
        }
    }
    __syncthreads();

    // ---- LN stats (rows of 256) ----
    {
        int r = t >> 4, q = t & 15;
        float s = 0.f, s2 = 0.f;
        #pragma unroll
        for (int u = 0; u < 16; ++u) { float z = sZ[r * DIM + q + u * 16]; s += z; s2 += z * z; }
        sPart[r][q][0] = s; sPart[r][q][1] = s2;
    }
    __syncthreads();
    if ((t & 15) == 0) {
        int r = t >> 4;
        float S = 0.f, S2 = 0.f;
        #pragma unroll
        for (int u = 0; u < 16; ++u) { S += sPart[r][u][0]; S2 += sPart[r][u][1]; }
        float mean = S * (1.f / DIM);
        float var  = S2 * (1.f / DIM) - mean * mean;
        sStats[r][0] = mean;
        sStats[r][1] = rsqrtf(var + LNEPS);
    }
    __syncthreads();
    // ---- apply LN + ReLU, convert to f16 for GEMM2 ----
    {
        int r = t >> 4, q = t & 15;
        float mean = sStats[r][0], rs = sStats[r][1];
        #pragma unroll
        for (int u = 0; u < 16; ++u) {
            int c = q + u * 16;
            float h = (sZ[r * DIM + c] - mean) * rs * g1[c] + be1[c];
            sH[r * DIM + c] = (_Float16)fmaxf(h, 0.f);
        }
    }
    __syncthreads();

    // ================= GEMM2: [16,256] x W2t -> z2 [16,256] =================
    v8f d0 = {0.f,0.f,0.f,0.f,0.f,0.f,0.f,0.f};
    v8f d1 = {0.f,0.f,0.f,0.f,0.f,0.f,0.f,0.f};
    #pragma unroll 2
    for (int kk = 0; kk < DIM / 32; ++kk) {
        if (kk + 1 < DIM / 32) {
            stage_wslice(W2t, DIM, kk + 1, sB + ((kk + 1) & 1) * SBN, t);
            wait_async_le4();
        } else {
            wait_async_0();
        }
        __syncthreads();
        const _Float16* sBk = sB + (kk & 1) * SBN;
        const _Float16* ap = &sH[mrow * DIM + kk * 32 + hi * 16];
        v8h alo = *(const v8h*)ap;
        v8h ahi = *(const v8h*)(ap + 8);
        v16h a = __builtin_shufflevector(alo, ahi, 0,1,2,3,4,5,6,7,8,9,10,11,12,13,14,15);
        v16h bf0 = *(const v16h*)(sBk + n0 * 32 + hi * 16);
        v16h bf1 = *(const v16h*)(sBk + n1 * 32 + hi * 16);
        d0 = __builtin_amdgcn_wmma_f32_16x16x32_f16(false, a, false, bf0, (short)0, d0, false, false);
        d1 = __builtin_amdgcn_wmma_f32_16x16x32_f16(false, a, false, bf1, (short)0, d1, false, false);
        __syncthreads();
    }
    {
        float bb0 = b2[n0], bb1 = b2[n1];
        #pragma unroll
        for (int r = 0; r < 8; ++r) {
            int row = r + 8 * hi;
            sZ[row * DIM + n0] = d0[r] + bb0;
            sZ[row * DIM + n1] = d1[r] + bb1;
        }
    }
    __syncthreads();

    // ---- LN stats for z2 ----
    {
        int r = t >> 4, q = t & 15;
        float s = 0.f, s2 = 0.f;
        #pragma unroll
        for (int u = 0; u < 16; ++u) { float z = sZ[r * DIM + q + u * 16]; s += z; s2 += z * z; }
        sPart[r][q][0] = s; sPart[r][q][1] = s2;
    }
    __syncthreads();
    if ((t & 15) == 0) {
        int r = t >> 4;
        float S = 0.f, S2 = 0.f;
        #pragma unroll
        for (int u = 0; u < 16; ++u) { S += sPart[r][u][0]; S2 += sPart[r][u][1]; }
        float mean = S * (1.f / DIM);
        float var  = S2 * (1.f / DIM) - mean * mean;
        sStats[r][0] = mean;
        sStats[r][1] = rsqrtf(var + LNEPS);
    }
    __syncthreads();

    // ---- LN + ReLU + column-sum over the 16 pairs -> pooled accumulator ----
    {
        float g2c = g2[t], be2c = be2[t];
        float cs = 0.f;
        #pragma unroll
        for (int r = 0; r < 16; ++r) {
            float h = (sZ[r * DIM + t] - sStats[r][0]) * sStats[r][1] * g2c + be2c;
            cs += fmaxf(h, 0.f);
        }
        unsafeAtomicAdd(&pooled[(size_t)b * DIM + t], cs);
    }
}

// ---------------- final head: pooled -> W3 -> LN -> ReLU ----------------
__global__ void k_final(const float* __restrict__ pooled,
                        const float* __restrict__ W3, const float* __restrict__ b3,
                        const float* __restrict__ g3, const float* __restrict__ be3,
                        float* __restrict__ out) {
    __shared__ float sP[DIM];
    __shared__ float sRed[OUTD];
    __shared__ float sPart[16];
    __shared__ float sStat[2];
    int b = blockIdx.x, t = threadIdx.x;
    sP[t] = pooled[(size_t)b * DIM + t] * (1.f / (float)NPAIR);
    __syncthreads();
    float acc = b3[t];
    for (int k = 0; k < DIM; ++k) acc += sP[k] * W3[(size_t)k * OUTD + t];
    sRed[t] = acc;
    __syncthreads();
    if (t < 16) { float s = 0.f; for (int u = 0; u < 16; ++u) s += sRed[t * 16 + u]; sPart[t] = s; }
    __syncthreads();
    if (t == 0) { float s = 0.f; for (int u = 0; u < 16; ++u) s += sPart[u]; sStat[0] = s * (1.f / OUTD); }
    __syncthreads();
    float mean = sStat[0];
    float d = acc - mean;
    sRed[t] = d * d;
    __syncthreads();
    if (t < 16) { float s = 0.f; for (int u = 0; u < 16; ++u) s += sRed[t * 16 + u]; sPart[t] = s; }
    __syncthreads();
    if (t == 0) { float s = 0.f; for (int u = 0; u < 16; ++u) s += sPart[u]; sStat[1] = rsqrtf(s * (1.f / OUTD) + LNEPS); }
    __syncthreads();
    out[(size_t)b * OUTD + t] = fmaxf(d * sStat[1] * g3[t] + be3[t], 0.f);
}

extern "C" void kernel_launch(void* const* d_in, const int* in_sizes, int n_in,
                              void* d_out, int out_size, void* d_ws, size_t ws_size,
                              hipStream_t stream) {
    const float* x   = (const float*)d_in[0];
    const float* W1  = (const float*)d_in[1];
    const float* b1  = (const float*)d_in[2];
    const float* g1  = (const float*)d_in[3];
    const float* be1 = (const float*)d_in[4];
    const float* W2  = (const float*)d_in[5];
    const float* b2  = (const float*)d_in[6];
    const float* g2  = (const float*)d_in[7];
    const float* be2 = (const float*)d_in[8];
    const float* W3  = (const float*)d_in[9];
    const float* b3  = (const float*)d_in[10];
    const float* g3  = (const float*)d_in[11];
    const float* be3 = (const float*)d_in[12];
    float* out = (float*)d_out;

    char* ws = (char*)d_ws;
    _Float16* xh  = (_Float16*)(ws);                                   // 2 MB
    _Float16* W1t = (_Float16*)(ws + (2u << 20));                      // 256 KB
    _Float16* W2t = (_Float16*)(ws + (2u << 20) + (256u << 10));       // 128 KB
    float*   pooled = (float*)(ws + (2u << 20) + (384u << 10));        // 64 KB

    int nx = BATCH * NOBJ * DIM;
    k_f2h<<<(nx + 255) / 256, 256, 0, stream>>>(x, xh, nx);
    k_transpose_h<<<(DIM2 * DIM + 255) / 256, 256, 0, stream>>>(W1, W1t, DIM2, DIM);
    k_transpose_h<<<(DIM * DIM + 255) / 256, 256, 0, stream>>>(W2, W2t, DIM, DIM);
    k_zero<<<(BATCH * DIM + 255) / 256, 256, 0, stream>>>(pooled, BATCH * DIM);
    k_pairs<<<BATCH * NBLK, 256, 0, stream>>>(xh, W1t, b1, g1, be1, W2t, b2, g2, be2, pooled);
    k_final<<<BATCH, 256, 0, stream>>>(pooled, W3, b3, g3, be3, out);
}
